// MultiHeadAttention_20641612825257
// MI455X (gfx1250) — matrix operand order repaired and checked by reading, not compiled, not run on hardware
//
#include <hip/hip_runtime.h>

typedef __attribute__((ext_vector_type(16))) _Float16 v16h;
typedef __attribute__((ext_vector_type(8)))  _Float16 v8h;
typedef __attribute__((ext_vector_type(8)))  float    v8f;
typedef __attribute__((ext_vector_type(4)))  float    v4f;

#define D_MODEL 1024
#define N_HEADS 16
#define D_K     64
#define BATCH   2
#define SEQ     2048
#define MROWS   (BATCH * SEQ)   // 4096

// ---------------------------------------------------------------------------
// Fragment loaders (all-f16 paths; contiguous k order, same for A and B).
// A-matrix (16x32 f16): lane m=L%16, half=L/16;
//   elems 0..7  -> K = kk + half*16 + e       (contiguous 8 halves, 16B)
//   elems 8..15 -> K = kk + half*16 + 8 + e-8 (contiguous 8 halves, 16B)
// B-matrix (32x16 f16): lane n=L%16, half=L/16;
//   elems 0..15 -> K = kk + half*16 + e       (contiguous 16 halves, 32B)
// ---------------------------------------------------------------------------

__device__ __forceinline__ v16h afrag_f16(const _Float16* rowp, int kk, int half) {
  v8h lo = *(const v8h*)(rowp + kk + half * 16);
  v8h hi = *(const v8h*)(rowp + kk + half * 16 + 8);
  v16h r;
#pragma unroll
  for (int i = 0; i < 8; ++i) { r[i] = lo[i]; r[8 + i] = hi[i]; }
  return r;
}

__device__ __forceinline__ v16h bfrag_f16(const _Float16* rowp, int kk, int half) {
  return *(const v16h*)(rowp + kk + half * 16);
}

#define WMMA_F16(A, B, C) \
  __builtin_amdgcn_wmma_f32_16x16x32_f16(false, (A), false, (B), (short)0, (C), false, false)

// Load one K-slice worth of fragments (A row + 4 B tiles) into a buffer set.
__device__ __forceinline__ void load_set(const _Float16* arow, const _Float16* brow0,
                                         int kk, int half, v16h& a, v16h b[4]) {
  a = afrag_f16(arow, kk, half);
#pragma unroll
  for (int t = 0; t < 4; ++t)
    b[t] = bfrag_f16(brow0 + (long)(16 * t) * D_MODEL, kk, half);
}

__device__ __forceinline__ void wmma_set(const v16h& a, const v16h b[4], v8f acc[4]) {
#pragma unroll
  for (int t = 0; t < 4; ++t)
    acc[t] = WMMA_F16(a, b[t], acc[t]);
}

// ---------------------------------------------------------------------------
// fp32 -> f16 pre-conversion (bandwidth-trivial; removes all cvt from GEMMs)
// ---------------------------------------------------------------------------
__global__ __launch_bounds__(256)
void cvt_f32_to_f16_kernel(const float* __restrict__ src, _Float16* __restrict__ dst,
                           int n) {
  const int i = (blockIdx.x * 256 + threadIdx.x) * 8;
  if (i >= n) return;
  v4f a = *(const v4f*)(src + i);
  v4f b = *(const v4f*)(src + i + 4);
  v8h r;
  r[0] = (_Float16)a[0]; r[1] = (_Float16)a[1];
  r[2] = (_Float16)a[2]; r[3] = (_Float16)a[3];
  r[4] = (_Float16)b[0]; r[5] = (_Float16)b[1];
  r[6] = (_Float16)b[2]; r[7] = (_Float16)b[3];
  *(v8h*)(dst + i) = r;
}

// ---------------------------------------------------------------------------
// Projection GEMM (all f16): Y[m,n] = sum_k X[m,k]*W[n,k], M=4096,N=1024,K=1024
// Wave tile 16x64, ping-pong double-buffered (unroll-2): no rotation copies,
// loads for slice i+1 in flight while WMMAs consume slice i.
// Output f16 scattered with strides (b,h,s,d) so V can be stored transposed.
// ---------------------------------------------------------------------------
__global__ __launch_bounds__(256)
void proj_gemm_kernel(const _Float16* __restrict__ X, const _Float16* __restrict__ W,
                      _Float16* __restrict__ Y,
                      long sB, long sH, long sS, long sD) {
  const int lane = threadIdx.x & 31;
  const int wave = threadIdx.x >> 5;
  const int m    = lane & 15;
  const int half = lane >> 4;
  const int rowTile = blockIdx.y * 128 + wave * 16;
  const int colTile = blockIdx.x * 64;

  const _Float16* arow  = X + (long)(rowTile + m) * D_MODEL;
  const _Float16* brow0 = W + (long)(colTile + m) * D_MODEL;

  v8f acc[4] = {};
  v16h a0, a1, b0[4], b1[4];

  load_set(arow, brow0, 0, half, a0, b0);
  for (int k0 = 0; k0 < D_MODEL - 64; k0 += 64) {
    load_set(arow, brow0, k0 + 32, half, a1, b1);
    __builtin_prefetch(arow + k0 + 192, 0, 3);   // global_prefetch_b8
    wmma_set(a0, b0, acc);
    load_set(arow, brow0, k0 + 64, half, a0, b0);
    wmma_set(a1, b1, acc);
  }
  // drain: buf0 holds slice K-64; load final slice K-32 into buf1
  load_set(arow, brow0, D_MODEL - 32, half, a1, b1);
  wmma_set(a0, b0, acc);
  wmma_set(a1, b1, acc);

#pragma unroll
  for (int t = 0; t < 4; ++t) {
    const int n  = colTile + 16 * t + m;
    const int hh = n >> 6;       // head
    const int dd = n & 63;       // dim within head
#pragma unroll
    for (int r = 0; r < 8; ++r) {
      const int mg = rowTile + r + 8 * half;   // global row (b*S + s)
      const int bb = mg >> 11;
      const int ss = mg & (SEQ - 1);
      Y[bb * sB + hh * sH + (long)ss * sS + (long)dd * sD] = (_Float16)acc[t][r];
    }
  }
}

// ---------------------------------------------------------------------------
// Output GEMM (f16 x f16 -> fp32 + bias): OUT = CTX @ Wo^T + bo
// Same ping-pong double-buffered structure.
// ---------------------------------------------------------------------------
__global__ __launch_bounds__(256)
void out_gemm_kernel(const _Float16* __restrict__ CTX, const _Float16* __restrict__ W,
                     const float* __restrict__ bias, float* __restrict__ OUT) {
  const int lane = threadIdx.x & 31;
  const int wave = threadIdx.x >> 5;
  const int m    = lane & 15;
  const int half = lane >> 4;
  const int rowTile = blockIdx.y * 128 + wave * 16;
  const int colTile = blockIdx.x * 64;

  const _Float16* arow  = CTX + (long)(rowTile + m) * D_MODEL;
  const _Float16* brow0 = W + (long)(colTile + m) * D_MODEL;

  v8f acc[4] = {};
  v16h a0, a1, b0[4], b1[4];

  load_set(arow, brow0, 0, half, a0, b0);
  for (int k0 = 0; k0 < D_MODEL - 64; k0 += 64) {
    load_set(arow, brow0, k0 + 32, half, a1, b1);
    __builtin_prefetch(arow + k0 + 192, 0, 3);
    wmma_set(a0, b0, acc);
    load_set(arow, brow0, k0 + 64, half, a0, b0);
    wmma_set(a1, b1, acc);
  }
  load_set(arow, brow0, D_MODEL - 32, half, a1, b1);
  wmma_set(a0, b0, acc);
  wmma_set(a1, b1, acc);

#pragma unroll
  for (int t = 0; t < 4; ++t) {
    const int n = colTile + 16 * t + m;
    const float bn2 = bias[n];
#pragma unroll
    for (int r = 0; r < 8; ++r) {
      const int mg = rowTile + r + 8 * half;
      OUT[(long)mg * D_MODEL + n] = acc[t][r] + bn2;
    }
  }
}

// ---------------------------------------------------------------------------
// Flash attention: one wave per (b, h, 16-query-row tile); online softmax
// over 32-key chunks; causal mask applied analytically.
//   QP, KP: [B,H,S,D_K] f16    VT: [B,H,D_K,S] f16    CTX: [B,S,H*D_K] f16
// ---------------------------------------------------------------------------
__global__ __launch_bounds__(256)
void attn_kernel(const _Float16* __restrict__ QP, const _Float16* __restrict__ KP,
                 const _Float16* __restrict__ VT, _Float16* __restrict__ CTX) {
  __shared__ _Float16 lsP[8][16 * 32];   // per-wave P staging (C-layout -> A-layout)

  const int lane = threadIdx.x & 31;
  const int wave = threadIdx.x >> 5;
  const int m    = lane & 15;
  const int half = lane >> 4;

  const int gid   = blockIdx.x * 8 + wave;  // 0 .. B*H*(S/16)-1
  const int qtile = gid & 127;              // S/16 = 128
  const int bh    = gid >> 7;               // b*16 + h
  const int qbase = qtile * 16;

  const _Float16* qrow = QP + ((long)bh * SEQ + qbase + m) * D_K;
  const v16h aq0 = afrag_f16(qrow, 0, half);
  const v16h aq1 = afrag_f16(qrow, 32, half);

  v8f acc[4] = {};
  float mstat[8], lstat[8];
#pragma unroll
  for (int r = 0; r < 8; ++r) { mstat[r] = -3.0e38f; lstat[r] = 0.0f; }

  const float scale = 0.125f;  // 1/sqrt(D_K)
  _Float16* lp = lsP[wave];

  const int kend = qbase + 15;
  for (int kc = 0; kc <= kend; kc += 32) {
    // ---- scores tile 16x32 = two 16x16 C tiles, K=64 in two WMMAs each ----
    const _Float16* kr0 = KP + ((long)bh * SEQ + kc + m) * D_K;
    const _Float16* kr1 = KP + ((long)bh * SEQ + kc + 16 + m) * D_K;
    v16h kb[4];
    kb[0] = bfrag_f16(kr0, 0, half);
    kb[1] = bfrag_f16(kr0, 32, half);
    kb[2] = bfrag_f16(kr1, 0, half);
    kb[3] = bfrag_f16(kr1, 32, half);

    v8f c0 = {}, c1 = {};
    c0 = WMMA_F16(aq0, kb[0], c0);
    c0 = WMMA_F16(aq1, kb[1], c0);
    c1 = WMMA_F16(aq0, kb[2], c1);
    c1 = WMMA_F16(aq1, kb[3], c1);

    // Prefetch the V chunk while softmax math runs.
    v16h vb[4];
#pragma unroll
    for (int t4 = 0; t4 < 4; ++t4) {
      const _Float16* vr = VT + ((long)bh * D_K + 16 * t4 + m) * SEQ + kc;
      vb[t4] = *(const v16h*)(vr + half * 16);
    }

    // ---- causal mask + online softmax (row stats across 16-lane groups) ---
#pragma unroll
    for (int r = 0; r < 8; ++r) {
      const int row  = qbase + r + 8 * half;
      const int col0 = kc + m;
      const int col1 = kc + 16 + m;
      float s0 = c0[r] * scale; if (col0 > row) s0 = -1.0e9f;
      float s1 = c1[r] * scale; if (col1 > row) s1 = -1.0e9f;

      float t = fmaxf(s0, s1);
      t = fmaxf(t, __shfl_xor(t, 1, 32));
      t = fmaxf(t, __shfl_xor(t, 2, 32));
      t = fmaxf(t, __shfl_xor(t, 4, 32));
      t = fmaxf(t, __shfl_xor(t, 8, 32));

      const float mnew  = fmaxf(mstat[r], t);
      const float alpha = __expf(mstat[r] - mnew);
      const float p0 = __expf(s0 - mnew);
      const float p1 = __expf(s1 - mnew);

      float rs = p0 + p1;
      rs += __shfl_xor(rs, 1, 32);
      rs += __shfl_xor(rs, 2, 32);
      rs += __shfl_xor(rs, 4, 32);
      rs += __shfl_xor(rs, 8, 32);

      lstat[r] = lstat[r] * alpha + rs;
      mstat[r] = mnew;
#pragma unroll
      for (int t4 = 0; t4 < 4; ++t4) acc[t4][r] *= alpha;

      const int lrow = r + 8 * half;
      lp[lrow * 32 + m]      = (_Float16)p0;
      lp[lrow * 32 + 16 + m] = (_Float16)p1;
    }

    // wave-private LDS: DS ops are in-order per wave; wait for stores to land
    asm volatile("s_wait_dscnt 0" ::: "memory");

    // ---- P (A-layout from LDS) @ V-chunk (B fragments already resident) ---
    const v16h pA = afrag_f16(lp + m * 32, 0, half);
#pragma unroll
    for (int t4 = 0; t4 < 4; ++t4)
      acc[t4] = WMMA_F16(pA, vb[t4], acc[t4]);
    asm volatile("" ::: "memory");  // keep next chunk's LDS stores after these loads
  }

  // ---- epilogue: normalize and scatter to ctx [B,S,H*D_K] ----
  const int hh = bh & 15;
  const int bb = bh >> 4;
#pragma unroll
  for (int t4 = 0; t4 < 4; ++t4) {
    const int col = hh * 64 + 16 * t4 + m;
#pragma unroll
    for (int r = 0; r < 8; ++r) {
      const int row = qbase + r + 8 * half;
      CTX[((long)bb * SEQ + row) * D_MODEL + col] = (_Float16)(acc[t4][r] / lstat[r]);
    }
  }
}

// ---------------------------------------------------------------------------
extern "C" void kernel_launch(void* const* d_in, const int* in_sizes, int n_in,
                              void* d_out, int out_size, void* d_ws, size_t ws_size,
                              hipStream_t stream) {
  (void)in_sizes; (void)n_in; (void)out_size; (void)ws_size;
  const float* Q  = (const float*)d_in[0];
  const float* K  = (const float*)d_in[1];
  const float* V  = (const float*)d_in[2];
  // d_in[3] = causal mask (bool) -- applied analytically in attn_kernel
  const float* Wq = (const float*)d_in[4];
  const float* Wk = (const float*)d_in[5];
  const float* Wv = (const float*)d_in[6];
  const float* Wo = (const float*)d_in[7];
  const float* bo = (const float*)d_in[8];

  const size_t PLANE  = (size_t)MROWS * D_MODEL * sizeof(_Float16);    // 8 MB
  const size_t WPLANE = (size_t)D_MODEL * D_MODEL * sizeof(_Float16);  // 2 MB
  char* ws = (char*)d_ws;
  _Float16* Qh  = (_Float16*)(ws + 0 * PLANE);
  _Float16* Kh  = (_Float16*)(ws + 1 * PLANE);
  _Float16* Vh  = (_Float16*)(ws + 2 * PLANE);
  _Float16* Wqh = (_Float16*)(ws + 3 * PLANE + 0 * WPLANE);
  _Float16* Wkh = (_Float16*)(ws + 3 * PLANE + 1 * WPLANE);
  _Float16* Wvh = (_Float16*)(ws + 3 * PLANE + 2 * WPLANE);
  _Float16* Woh = (_Float16*)(ws + 3 * PLANE + 3 * WPLANE);
  _Float16* qp  = (_Float16*)(ws + 4 * PLANE);
  _Float16* kp  = (_Float16*)(ws + 5 * PLANE);
  _Float16* vT  = (_Float16*)(ws + 6 * PLANE);
  _Float16* ctx = (_Float16*)(ws + 7 * PLANE);

  const int NA = MROWS * D_MODEL;     // 4M elements
  const int NW = D_MODEL * D_MODEL;   // 1M elements
  cvt_f32_to_f16_kernel<<<NA / (256 * 8), 256, 0, stream>>>(Q,  Qh,  NA);
  cvt_f32_to_f16_kernel<<<NA / (256 * 8), 256, 0, stream>>>(K,  Kh,  NA);
  cvt_f32_to_f16_kernel<<<NA / (256 * 8), 256, 0, stream>>>(V,  Vh,  NA);
  cvt_f32_to_f16_kernel<<<NW / (256 * 8), 256, 0, stream>>>(Wq, Wqh, NW);
  cvt_f32_to_f16_kernel<<<NW / (256 * 8), 256, 0, stream>>>(Wk, Wkh, NW);
  cvt_f32_to_f16_kernel<<<NW / (256 * 8), 256, 0, stream>>>(Wv, Wvh, NW);
  cvt_f32_to_f16_kernel<<<NW / (256 * 8), 256, 0, stream>>>(Wo, Woh, NW);

  const long sB = (long)N_HEADS * SEQ * D_K;   // 2097152
  const long sH = (long)SEQ * D_K;             // 131072

  dim3 gGemm(D_MODEL / 64, MROWS / 128);       // (16, 32)
  proj_gemm_kernel<<<gGemm, 256, 0, stream>>>(Qh, Wqh, qp, sB, sH, D_K, 1);
  proj_gemm_kernel<<<gGemm, 256, 0, stream>>>(Kh, Wkh, kp, sB, sH, D_K, 1);
  proj_gemm_kernel<<<gGemm, 256, 0, stream>>>(Vh, Wvh, vT, sB, sH, 1, SEQ);  // transposed

  const int nWaveTiles = BATCH * N_HEADS * (SEQ / 16);  // 4096
  attn_kernel<<<nWaveTiles / 8, 256, 0, stream>>>(qp, kp, vT, ctx);

  out_gemm_kernel<<<gGemm, 256, 0, stream>>>(ctx, Woh, bo, (float*)d_out);
}
